// CrossModalAttention_70841190580455
// MI455X (gfx1250) — matrix operand order repaired and checked by reading, not compiled, not run on hardware
//
#include <hip/hip_runtime.h>

// ---------------------------------------------------------------------------
// CDNA5 (gfx1250) wave32 WMMA helpers: v_wmma_f32_16x16x32_bf16
// ---------------------------------------------------------------------------
typedef __attribute__((ext_vector_type(16))) __bf16 v16bf;
typedef __attribute__((ext_vector_type(4)))  __bf16 v4bf;
typedef __attribute__((ext_vector_type(8)))  float  v8f;
typedef __attribute__((ext_vector_type(4)))  unsigned int v4u;
typedef __attribute__((ext_vector_type(8)))  int v8i;
typedef __attribute__((ext_vector_type(4)))  int v4i;

union Frag16 {
  v16bf v;
  int4  q[2];
};

// A-matrix 16x32 (MxK), bf16, row-major source, leading dim `ld`.
// ISA layout: lane m = lane&15; K-halves 0..7/16..23 (lanes<16), 8..15/24..31.
__device__ __forceinline__ v16bf load_frag_a16x32(const __bf16* p, int ld) {
  const int lane = threadIdx.x & 31;
  const int m = lane & 15;
  const int kh = lane >> 4;
  const __bf16* r = p + (size_t)m * ld;
  Frag16 f;
  f.q[0] = *(const int4*)(r + kh * 8);
  f.q[1] = *(const int4*)(r + 16 + kh * 8);
  return f.v;
}

// B-matrix 32x16 (KxN) from an N-major source (B[k][n] = src[n][k]).
// ISA layout: lane col n = lane&15; K-halves 0..7/16..23 (lanes<16), 8..15/24..31.
__device__ __forceinline__ v16bf load_frag_bT32x16(const __bf16* p, int ld) {
  const int lane = threadIdx.x & 31;
  const int n = lane & 15;
  const int kh = lane >> 4;
  const __bf16* r = p + (size_t)n * ld;
  Frag16 f;
  f.q[0] = *(const int4*)(r + kh * 8);
  f.q[1] = *(const int4*)(r + 16 + kh * 8);
  return f.v;
}

__device__ __forceinline__ v8f wmma_bf16(v16bf a, v16bf b, v8f c) {
  return __builtin_amdgcn_wmma_f32_16x16x32_bf16(false, a, false, b, (short)0, c,
                                                 false, false);
}

// ---------------------------------------------------------------------------
// Tensor Data Mover staging (gfx1250): 2D tile -> LDS, with hardware row
// padding (64B data + 16B pad == our 40-element LDS row pitch).
// This toolchain exposes the 6-arg builtin:
//   (uint32x4 g0, int32x8 g1, int32x4, int32x4, int32x8, i32 cpol)
// ---------------------------------------------------------------------------
#if defined(__gfx1250__) && __has_builtin(__builtin_amdgcn_tensor_load_to_lds)
#define USE_TDM 1
#else
#define USE_TDM 0
#endif

#if USE_TDM
// Flat shared pointers carry the LDS byte offset in their low 32 bits
// (aperture base occupies [63:32] only).
__device__ __forceinline__ unsigned lds_off(const void* p) {
  return (unsigned)(unsigned long long)p;
}

// Load a rows x 32 bf16 tile (row stride `ld` elems) starting at gsrc into LDS
// at byte offset lds_addr, padded to 80B row pitch (pad_interval=16DW -> +4DW).
__device__ __forceinline__ void tdm_load_tile(const __bf16* gsrc,
                                              unsigned lds_addr, int rows,
                                              int ld, int tensor_rows) {
  const unsigned long long ga = (unsigned long long)gsrc;
  v4u g0;
  g0[0] = 1u;                                   // count=1, user descriptor
  g0[1] = lds_addr;                             // LDS byte address
  g0[2] = (unsigned)(ga & 0xFFFFFFFFu);         // global_addr[31:0]
  g0[3] = (unsigned)((ga >> 32) & 0x1FFFFFFu)   // global_addr[56:32]
          | (2u << 30);                         // type=2 ("image")
  v8i g1;
  g1[0] = (int)((1u << 16)      // data_size = 2 bytes
                | (1u << 20)    // pad_enable
                | (3u << 22)    // pad_interval: every 16 DWORDs (64B row)
                | (3u << 25));  // pad_amount: 4 DWORDs (16B)
  g1[1] = (int)(((unsigned)ld & 0xFFFFu) << 16);               // tensor_dim0 lo
  g1[2] = (int)((((unsigned)ld >> 16) & 0xFFFFu) |
                (((unsigned)tensor_rows & 0xFFFFu) << 16));    // dim0 hi|dim1 lo
  g1[3] = (int)((((unsigned)tensor_rows >> 16) & 0xFFFFu) |
                (32u << 16));                                  // tile_dim0 = 32
  g1[4] = rows;                                                // tile_dim1
  g1[5] = ld;                                                  // dim0_stride lo
  g1[6] = 0;
  g1[7] = 0;
  const v4i z4 = {0, 0, 0, 0};                    // groups 2/3 unused (2D tile)
  const v8i z8 = {0, 0, 0, 0, 0, 0, 0, 0};        // extra group (clang-23 form)
  __builtin_amdgcn_tensor_load_to_lds(g0, g1, z4, z4, z8, 0);
}
#endif

// ---------------------------------------------------------------------------
// f32 -> bf16 conversion (grid-stride)
// ---------------------------------------------------------------------------
__global__ __launch_bounds__(256) void cma_cvt_bf16(const float* __restrict__ x,
                                                    __bf16* __restrict__ y,
                                                    long n) {
  long i = (long)blockIdx.x * 256 + threadIdx.x;
  const long stride = (long)gridDim.x * 256;
  for (; i < n; i += stride) y[i] = (__bf16)x[i];
}

// ---------------------------------------------------------------------------
// Double-buffered WMMA GEMM: C[M,N] = alpha * (A[M,K] * W[N,K]^T + bias)
// Block tile 64x128x32, 256 threads = 8 waves as 2(M) x 4(N); each wave owns a
// 32x32 C tile = 4 v_wmma accumulators -> 4 WMMA per single barrier per k-step.
// Tile staging: TDM tensor_load_to_lds (wave 0 issues, s_wait_tensorcnt) with
// cooperative-load fallback. MODE 0: bf16 [M,N]; 1: bf16 Vt[b,n,key]; 2: f32 +
// residual.
// ---------------------------------------------------------------------------
constexpr int BM = 64, BN = 128, BK = 32, LDP = BK + 8;  // 80B rows, 16B align

template <int MODE>
__global__ __launch_bounds__(256) void cma_gemm_wmma(
    const __bf16* __restrict__ A, const __bf16* __restrict__ W,
    const float* __restrict__ bias, const float* __restrict__ resid,
    void* __restrict__ Out, int M, int N, int K, float alpha, int LkvT) {
  __shared__ __bf16 As[2][BM][LDP];
  __shared__ __bf16 Ws[2][BN][LDP];

  const int m0 = blockIdx.x * BM;
  const int n0 = blockIdx.y * BN;
  const int tid = threadIdx.x;
  const int wid = tid >> 5, lane = tid & 31;
  const int wm = wid & 1, wn = wid >> 1;  // 2 x 4 wave grid

  v8f c00 = {}, c01 = {}, c10 = {}, c11 = {};
  const int kSteps = K / BK;
  int buf = 0;

#if USE_TDM
  if (wid == 0) {
    tdm_load_tile(&A[(size_t)m0 * K], lds_off(&As[0][0][0]), BM, K, M - m0);
    tdm_load_tile(&W[(size_t)n0 * K], lds_off(&Ws[0][0][0]), BN, K, N - n0);
    __builtin_amdgcn_s_wait_tensorcnt(0);
  }
  __syncthreads();
#else
  const int arow = tid >> 2, acol = (tid & 3) * 8;   // 64x32: 1 int4/thread
  const int wrow = tid >> 1, wcol = (tid & 1) * 16;  // 128x32: 2 int4/thread
  *(int4*)&As[0][arow][acol] = *(const int4*)&A[(size_t)(m0 + arow) * K + acol];
  *(int4*)&Ws[0][wrow][wcol] = *(const int4*)&W[(size_t)(n0 + wrow) * K + wcol];
  *(int4*)&Ws[0][wrow][wcol + 8] =
      *(const int4*)&W[(size_t)(n0 + wrow) * K + wcol + 8];
  __syncthreads();
#endif

  for (int ks = 0; ks < kSteps; ++ks) {
    const int kn = (ks + 1) * BK;  // next tile
#if USE_TDM
    if (wid == 0 && ks + 1 < kSteps) {  // DMA next tiles while waves compute
      tdm_load_tile(&A[(size_t)m0 * K + kn], lds_off(&As[buf ^ 1][0][0]), BM, K,
                    M - m0);
      tdm_load_tile(&W[(size_t)n0 * K + kn], lds_off(&Ws[buf ^ 1][0][0]), BN, K,
                    N - n0);
    }
#else
    if (ks + 1 < kSteps) {  // stage next tiles into the other buffer
      *(int4*)&As[buf ^ 1][arow][acol] =
          *(const int4*)&A[(size_t)(m0 + arow) * K + kn + acol];
      *(int4*)&Ws[buf ^ 1][wrow][wcol] =
          *(const int4*)&W[(size_t)(n0 + wrow) * K + kn + wcol];
      *(int4*)&Ws[buf ^ 1][wrow][wcol + 8] =
          *(const int4*)&W[(size_t)(n0 + wrow) * K + kn + wcol + 8];
      if (kn + BK < K) {  // global_prefetch_b8 for tile after next
        __builtin_prefetch(&A[(size_t)(m0 + arow) * K + kn + BK + acol], 0, 0);
        __builtin_prefetch(&W[(size_t)(n0 + wrow) * K + kn + BK + wcol], 0, 0);
      }
    }
#endif
    const __bf16* Ab = &As[buf][wm * 32][0];
    const __bf16* Wb = &Ws[buf][wn * 32][0];
    v16bf a0 = load_frag_a16x32(Ab, LDP);
    v16bf a1 = load_frag_a16x32(Ab + 16 * LDP, LDP);
    v16bf b0 = load_frag_bT32x16(Wb, LDP);
    v16bf b1 = load_frag_bT32x16(Wb + 16 * LDP, LDP);
    c00 = wmma_bf16(a0, b0, c00);
    c01 = wmma_bf16(a0, b1, c01);
    c10 = wmma_bf16(a1, b0, c10);
    c11 = wmma_bf16(a1, b1, c11);
#if USE_TDM
    if (wid == 0 && ks + 1 < kSteps) __builtin_amdgcn_s_wait_tensorcnt(0);
#endif
    __syncthreads();
    buf ^= 1;
  }

  // C/D layout: lane col = lane&15; M = (lane>>4)*8 + vgpr index
  const int cn = lane & 15;
  const int mb = (lane >> 4) * 8;
  const v8f cc_[2][2] = {{c00, c01}, {c10, c11}};
#pragma unroll
  for (int i = 0; i < 2; ++i)
#pragma unroll
    for (int j = 0; j < 2; ++j)
#pragma unroll
      for (int r = 0; r < 8; ++r) {
        const int m = m0 + wm * 32 + i * 16 + mb + r;
        const int n = n0 + wn * 32 + j * 16 + cn;
        const float val = alpha * (cc_[i][j][r] + bias[n]);
        if (MODE == 0) {
          ((__bf16*)Out)[(size_t)m * N + n] = (__bf16)val;
        } else if (MODE == 1) {
          const int bb = m / LkvT, key = m % LkvT;
          ((__bf16*)Out)[((size_t)bb * N + n) * LkvT + key] = (__bf16)val;
        } else {
          ((float*)Out)[(size_t)m * N + n] = val + resid[(size_t)m * N + n];
        }
      }
}

// ---------------------------------------------------------------------------
// Attention: one block per (batch, 16-query tile); loops over 16 heads.
// Q/K/V panels (40MB bf16) are L2-resident (192MB L2) -> WMMA fragments are
// loaded straight from global. Scores -> LDS f32, exact softmax (float4
// vectorized), P as bf16 in LDS, ctx = P @ V via WMMA (4 N-tiles x 2 K-halves
// over the 8 waves). Head-averaged attn_weights accumulated block-locally.
// ---------------------------------------------------------------------------
constexpr int AB = 8, AH = 16, ALq = 512, ALkv = 1024, AD = 1024, Ahd = 64,
              QT = 16;

__global__ __launch_bounds__(256) void cma_attn_wmma(
    const __bf16* __restrict__ Qp,  // [B,Lq,D]  (pre-scaled by 1/sqrt(hd))
    const __bf16* __restrict__ Kp,  // [B,Lkv,D]
    const __bf16* __restrict__ Vt,  // [B,D,Lkv] (per-batch transposed)
    __bf16* __restrict__ Ctx,       // [B,Lq,D]
    float* __restrict__ AttnW) {    // [B,Lq,Lkv]
  __shared__ float  S[QT][ALkv];       // 64 KB scores / exp()
  __shared__ __bf16 P[QT][ALkv];       // 32 KB probabilities (WMMA A operand)
  __shared__ float  redm[QT][16];
  __shared__ float  reds[QT][16];
  __shared__ float  ctxp[2][QT][Ahd];  // K-split partial contexts

  const int b = blockIdx.x / (ALq / QT);
  const int q0 = (blockIdx.x % (ALq / QT)) * QT;
  const int tid = threadIdx.x;
  const int wid = tid >> 5, lane = tid & 31;
  const int row = tid >> 4, g = tid & 15;  // softmax: 16 threads x 64 cols
  const float invH = 1.0f / AH;

  for (int h = 0; h < AH; ++h) {
    const __bf16* Qbase = Qp + ((size_t)(b * ALq + q0)) * AD + h * Ahd;

    // ---- scores S = Qh (16x64) @ Kh^T (64x1024); Q frags are loop-invariant
    {
      const v16bf aq0 = load_frag_a16x32(Qbase, AD);
      const v16bf aq1 = load_frag_a16x32(Qbase + 32, AD);
      const int cn = lane & 15, mb = (lane >> 4) * 8;
      for (int t = wid; t < ALkv / 16; t += 8) {
        const __bf16* Kb = Kp + ((size_t)(b * ALkv + t * 16)) * AD + h * Ahd;
        v8f c = {};
        c = wmma_bf16(aq0, load_frag_bT32x16(Kb, AD), c);
        c = wmma_bf16(aq1, load_frag_bT32x16(Kb + 32, AD), c);
#pragma unroll
        for (int r = 0; r < 8; ++r) S[mb + r][t * 16 + cn] = c[r];
      }
    }
    __syncthreads();

    // ---- exact softmax over 1024 keys, float4-vectorized
    float4* Srow4 = (float4*)&S[row][0];
    const int cb = g * 16;  // 16 float4 per thread
    float mx = -3.0e38f;
#pragma unroll 4
    for (int cc = 0; cc < 16; ++cc) {
      const float4 v = Srow4[cb + cc];
      mx = fmaxf(mx, fmaxf(fmaxf(v.x, v.y), fmaxf(v.z, v.w)));
    }
    redm[row][g] = mx;
    __syncthreads();
    float m = redm[row][0];
#pragma unroll
    for (int i = 1; i < 16; ++i) m = fmaxf(m, redm[row][i]);
    float sum = 0.f;
#pragma unroll 4
    for (int cc = 0; cc < 16; ++cc) {
      float4 v = Srow4[cb + cc];
      v.x = __expf(v.x - m);
      v.y = __expf(v.y - m);
      v.z = __expf(v.z - m);
      v.w = __expf(v.w - m);
      Srow4[cb + cc] = v;
      sum += (v.x + v.y) + (v.z + v.w);
    }
    reds[row][g] = sum;
    __syncthreads();
    float tot = 0.f;
#pragma unroll
    for (int i = 0; i < 16; ++i) tot += reds[row][i];
    const float inv = 1.0f / tot;
    float4* wrow4 =
        (float4*)(AttnW + ((size_t)(b * ALq + q0 + row)) * ALkv);
#pragma unroll 4
    for (int cc = 0; cc < 16; ++cc) {
      const float4 e = Srow4[cb + cc];
      const float px = e.x * inv, py = e.y * inv, pz = e.z * inv,
                  pw = e.w * inv;
      v4bf pb;
      pb[0] = (__bf16)px; pb[1] = (__bf16)py;
      pb[2] = (__bf16)pz; pb[3] = (__bf16)pw;
      *(v4bf*)&P[row][(cb + cc) * 4] = pb;
      float4 acc = (h == 0) ? make_float4(0.f, 0.f, 0.f, 0.f)
                            : wrow4[cb + cc];  // block-owned: no race
      acc.x += px * invH; acc.y += py * invH;
      acc.z += pz * invH; acc.w += pw * invH;
      wrow4[cb + cc] = acc;
    }
    __syncthreads();

    // ---- ctx = P (16x1024) @ Vh (1024x64); wave = (N-tile, K-half)
    {
      const int t = wid & 3, half = wid >> 2;
      v8f c = {};
      const __bf16* Vb0 = Vt + ((size_t)b * AD + h * Ahd + t * 16) * ALkv;
      for (int kc = half * 512; kc < half * 512 + 512; kc += 32) {
        v16bf a = load_frag_a16x32(&P[0][kc], ALkv);
        v16bf bb = load_frag_bT32x16(Vb0 + kc, ALkv);
        c = wmma_bf16(a, bb, c);
      }
      const int cn = lane & 15, mb = (lane >> 4) * 8;
#pragma unroll
      for (int r = 0; r < 8; ++r) ctxp[half][mb + r][t * 16 + cn] = c[r];
    }
    __syncthreads();
    for (int i = tid; i < QT * Ahd; i += 256) {
      const int r = i >> 6, cN = i & 63;
      const float v = ctxp[0][r][cN] + ctxp[1][r][cN];
      Ctx[((size_t)(b * ALq + q0 + r)) * AD + h * Ahd + cN] = (__bf16)v;
    }
    __syncthreads();
  }
}

// ---------------------------------------------------------------------------
// Row-wise LayerNorm over D=1024 (input already contains residual + bias)
// ---------------------------------------------------------------------------
__global__ __launch_bounds__(256) void cma_layernorm(
    const float* __restrict__ X, const float* __restrict__ gamma,
    const float* __restrict__ beta, float* __restrict__ out, int Dn) {
  __shared__ float s1[256];
  __shared__ float s2[256];
  const float* x = X + (size_t)blockIdx.x * Dn;
  float a = 0.f, bq = 0.f;
  for (int i = threadIdx.x; i < Dn; i += 256) {
    const float v = x[i];
    a += v;
    bq += v * v;
  }
  s1[threadIdx.x] = a;
  s2[threadIdx.x] = bq;
  __syncthreads();
  for (int o = 128; o > 0; o >>= 1) {
    if (threadIdx.x < o) {
      s1[threadIdx.x] += s1[threadIdx.x + o];
      s2[threadIdx.x] += s2[threadIdx.x + o];
    }
    __syncthreads();
  }
  const float mu = s1[0] / Dn;
  const float var = s2[0] / Dn - mu * mu;
  const float rinv = rsqrtf(var + 1e-5f);
  for (int i = threadIdx.x; i < Dn; i += 256)
    out[(size_t)blockIdx.x * Dn + i] = (x[i] - mu) * rinv * gamma[i] + beta[i];
}

// ---------------------------------------------------------------------------
// Host-side launch
// ---------------------------------------------------------------------------
extern "C" void kernel_launch(void* const* d_in, const int* in_sizes, int n_in,
                              void* d_out, int out_size, void* d_ws,
                              size_t ws_size, hipStream_t stream) {
  (void)in_sizes; (void)n_in; (void)out_size; (void)ws_size;
  const float* query = (const float*)d_in[0];
  const float* key_  = (const float*)d_in[1];
  const float* value = (const float*)d_in[2];
  const float* in_w  = (const float*)d_in[3];   // [3D, D]
  const float* in_b  = (const float*)d_in[4];   // [3D]
  const float* out_w = (const float*)d_in[5];   // [D, D]
  const float* out_b = (const float*)d_in[6];   // [D]
  const float* gamma = (const float*)d_in[7];
  const float* beta  = (const float*)d_in[8];
  float* out = (float*)d_out;

  constexpr size_t Dd = 1024, MQ = 8 * 512, MK = 8 * 1024;
  char* ws = (char*)d_ws;
  size_t off = 0;
  auto alloc = [&](size_t bytes) {
    void* p = ws + off;
    off += (bytes + 255) & ~(size_t)255;
    return p;
  };
  __bf16* qx   = (__bf16*)alloc(MQ * Dd * 2);      // query bf16
  __bf16* kx   = (__bf16*)alloc(MK * Dd * 2);      // key bf16
  __bf16* vx   = (__bf16*)alloc(MK * Dd * 2);      // value bf16
  __bf16* wqkv = (__bf16*)alloc(3 * Dd * Dd * 2);  // in_proj_w bf16
  __bf16* wob  = (__bf16*)alloc(Dd * Dd * 2);      // out_proj_w bf16
  __bf16* Qp   = (__bf16*)alloc(MQ * Dd * 2);      // scaled Q projection
  __bf16* Kp   = (__bf16*)alloc(MK * Dd * 2);      // K projection
  __bf16* Vt   = (__bf16*)alloc(MK * Dd * 2);      // V projection, [B,D,Lkv]
  __bf16* Ctx  = (__bf16*)alloc(MQ * Dd * 2);      // attention context
  float*  AO   = (float*)alloc(MQ * Dd * 4);       // out-proj + residual (f32)

  cma_cvt_bf16<<<2048, 256, 0, stream>>>(query, qx, (long)(MQ * Dd));
  cma_cvt_bf16<<<2048, 256, 0, stream>>>(key_, kx, (long)(MK * Dd));
  cma_cvt_bf16<<<2048, 256, 0, stream>>>(value, vx, (long)(MK * Dd));
  cma_cvt_bf16<<<2048, 256, 0, stream>>>(in_w, wqkv, (long)(3 * Dd * Dd));
  cma_cvt_bf16<<<2048, 256, 0, stream>>>(out_w, wob, (long)(Dd * Dd));

  const float qscale = 0.125f;  // 1/sqrt(hd=64)
  cma_gemm_wmma<0><<<dim3(MQ / BM, Dd / BN), 256, 0, stream>>>(
      qx, wqkv, in_b, nullptr, Qp, (int)MQ, (int)Dd, (int)Dd, qscale, 0);
  cma_gemm_wmma<0><<<dim3(MK / BM, Dd / BN), 256, 0, stream>>>(
      kx, wqkv + Dd * Dd, in_b + Dd, nullptr, Kp, (int)MK, (int)Dd, (int)Dd,
      1.0f, 0);
  cma_gemm_wmma<1><<<dim3(MK / BM, Dd / BN), 256, 0, stream>>>(
      vx, wqkv + 2 * Dd * Dd, in_b + 2 * Dd, nullptr, Vt, (int)MK, (int)Dd,
      (int)Dd, 1.0f, 1024);

  float* attnW = out + MQ * Dd;  // second tuple output [B,Lq,Lkv]
  cma_attn_wmma<<<AB * (ALq / QT), 256, 0, stream>>>(Qp, Kp, Vt, Ctx, attnW);

  cma_gemm_wmma<2><<<dim3(MQ / BM, Dd / BN), 256, 0, stream>>>(
      Ctx, wob, out_b, query, AO, (int)MQ, (int)Dd, (int)Dd, 1.0f, 0);

  cma_layernorm<<<(int)MQ, 256, 0, stream>>>(AO, gamma, beta, out, (int)Dd);
}
